// SAGE_53266184405049
// MI455X (gfx1250) — matrix operand order repaired and checked by reading, not compiled, not run on hardware
//
#include <hip/hip_runtime.h>
#include <hip/hip_bf16.h>
#include <math.h>

#define N_NODES 50000
#define N_EDGES 800000
#define F 128
#define NUM_CLASSES 2

typedef __attribute__((ext_vector_type(16))) __bf16 v16bf;
typedef __attribute__((ext_vector_type(8)))  __bf16 v8bf;
typedef __attribute__((ext_vector_type(8)))  float  v8f;

// float -> bf16 via the compiler's native conversion: on gfx1250 this lowers
// to the hardware packed converter (v_cvt_pk_bf16_f32) instead of a manual
// bfe/add3/mov bit-twiddle chain.
__device__ __forceinline__ __bf16 f2bf(float f) {
    return static_cast<__bf16>(f);
}

// ---------------------------------------------------------------------------
// Kernel 1: zero the accumulators (summed[N*F] and deg[N]) in workspace.
// ---------------------------------------------------------------------------
__global__ void sage_zero_ws(float* __restrict__ ws, int n) {
    int stride = gridDim.x * blockDim.x;
    for (int i = blockIdx.x * blockDim.x + threadIdx.x; i < n; i += stride)
        ws[i] = 0.0f;
}

// ---------------------------------------------------------------------------
// Kernel 1b: one-shot conversion of W_self | W_neigh (row-major 128x128 f32)
// to row-major bf16 in workspace. 32K elements total; trivial cost.
// ---------------------------------------------------------------------------
__global__ void sage_convert_weights(const float* __restrict__ W_self,
                                     const float* __restrict__ W_neigh,
                                     __bf16* __restrict__ wbf) {
    int stride = gridDim.x * blockDim.x;
    for (int i = blockIdx.x * blockDim.x + threadIdx.x; i < 2 * F * F; i += stride) {
        float v = (i < F * F) ? W_self[i] : W_neigh[i - F * F];
        wbf[i] = f2bf(v);
    }
}

// ---------------------------------------------------------------------------
// Kernel 2: edge scatter. One wave32 per edge: lanes cooperatively atomicAdd
// one 128-float source row into the destination accumulator (L2-resident).
// ---------------------------------------------------------------------------
__global__ void __launch_bounds__(256)
sage_edge_scatter(const float*      __restrict__ in_feat,
                  const long long*  __restrict__ src,
                  const long long*  __restrict__ dst,
                  float*            __restrict__ summed,
                  float*            __restrict__ deg) {
    const int lane   = threadIdx.x & 31;
    const int wave   = (blockIdx.x * blockDim.x + threadIdx.x) >> 5;
    const int nwaves = (gridDim.x * blockDim.x) >> 5;

    for (int e = wave; e < N_EDGES; e += nwaves) {
        const int s = (int)src[e];
        const int d = (int)dst[e];

        // prefetch next edge's source row (emits global_prefetch_b8)
        const int en = e + nwaves;
        if (en < N_EDGES) {
            const char* p = (const char*)(in_feat + (long)((int)src[en]) * F);
            __builtin_prefetch(p + lane * 16, 0, 0);
        }

        const float* row = in_feat + (long)s * F;
        float*       acc = summed  + (long)d * F;
        #pragma unroll
        for (int j = 0; j < F / 32; ++j)
            atomicAdd(&acc[lane + 32 * j], row[lane + 32 * j]);
        if (lane == 0)
            atomicAdd(&deg[d], 1.0f);
    }
}

// ---------------------------------------------------------------------------
// Kernel 3: fused  h = relu(X @ Ws^T + (summed/deg) @ Wn^T + b)  then
//           out = sigmoid(h @ Wfc^T + bfc).
// One block = 16 node rows. 8 waves; wave w computes the 16x16 output tile
// for columns [16w, 16w+16) via v_wmma_f32_16x16x32_bf16 (K=128 in 4 chunks,
// two weight matrices -> 8 WMMAs per wave). B fragments come pre-converted
// from the bf16 weight copy (2x global_load_b128 per fragment, no VALU).
// h tile staged in LDS, then the 2-class FC + sigmoid done by 32 threads.
// ---------------------------------------------------------------------------
__global__ void __launch_bounds__(256)
sage_gemm_fused(const float*  __restrict__ in_feat,
                const float*  __restrict__ summed,
                const float*  __restrict__ deg,
                const __bf16* __restrict__ wbf,     // [Ws bf16 | Wn bf16]
                const float*  __restrict__ b_neigh,
                const float*  __restrict__ W_fc,
                const float*  __restrict__ b_fc,
                float*        __restrict__ out) {
    __shared__ float h_lds[16][F + 4];

    const int lane = threadIdx.x & 31;
    const int wv   = threadIdx.x >> 5;     // 0..7 : which 16-col n-tile
    const int half = lane >> 4;            // 0/1  : lane group
    const int l16  = lane & 15;
    const int m    = blockIdx.x * 16 + l16;  // global node row (A-matrix row)
    const int n    = wv * 16 + l16;          // output column 0..127 (B col)

    const float invdeg = __builtin_amdgcn_rcpf(fmaxf(deg[m], 1.0f));
    const float* xrow = in_feat + (long)m * F;
    const float* srow = summed  + (long)m * F;
    const __bf16* wsrow = wbf + n * F;            // W_self row n (bf16)
    const __bf16* wnrow = wbf + F * F + n * F;    // W_neigh row n (bf16)

    v8f acc = {};
    #pragma unroll
    for (int kc = 0; kc < 4; ++kc) {
        // A tiles: 16x32 bf16, contiguous k order matching B: k = 2v+16*half
        v16bf a_x, a_h;
        #pragma unroll
        for (int v = 0; v < 8; ++v) {
            const int k = 32 * kc + (v << 1) + (half << 4);
            a_x[2 * v]     = f2bf(xrow[k]);
            a_x[2 * v + 1] = f2bf(xrow[k + 1]);
            a_h[2 * v]     = f2bf(srow[k]     * invdeg);
            a_h[2 * v + 1] = f2bf(srow[k + 1] * invdeg);
        }
        // B tiles: 32x16 bf16, B[k,n] = W[n,k]; lane reads 16 contiguous bf16
        // (32 bytes = two b128 loads) from the pre-converted weight copy.
        const int kb = 32 * kc + (half << 4);
        const v8bf* ps = (const v8bf*)(wsrow + kb);
        const v8bf* pn = (const v8bf*)(wnrow + kb);
        const v8bf s0 = ps[0], s1 = ps[1];
        const v8bf n0 = pn[0], n1 = pn[1];
        const v16bf b_s = __builtin_shufflevector(
            s0, s1, 0, 1, 2, 3, 4, 5, 6, 7, 8, 9, 10, 11, 12, 13, 14, 15);
        const v16bf b_n = __builtin_shufflevector(
            n0, n1, 0, 1, 2, 3, 4, 5, 6, 7, 8, 9, 10, 11, 12, 13, 14, 15);

        acc = __builtin_amdgcn_wmma_f32_16x16x32_bf16(
                  false, a_x, false, b_s, (short)0, acc, false, false);
        acc = __builtin_amdgcn_wmma_f32_16x16x32_bf16(
                  false, a_h, false, b_n, (short)0, acc, false, false);
    }

    // bias + relu, stage to LDS. C/D layout: VGPR r -> row r + 8*half, col l16.
    const float bias = b_neigh[n];
    #pragma unroll
    for (int r = 0; r < 8; ++r) {
        const float h = acc[r] + bias;
        h_lds[r + 8 * half][n] = fmaxf(h, 0.0f);
    }
    __syncthreads();

    // Final FC (2 classes) + sigmoid: 32 threads, one (row, class) each.
    if (threadIdx.x < 32) {
        const int row = threadIdx.x >> 1;
        const int cls = threadIdx.x & 1;
        float s = b_fc[cls];
        #pragma unroll 8
        for (int k = 0; k < F; ++k)
            s += h_lds[row][k] * W_fc[cls * F + k];
        const float sig = 1.0f / (1.0f + __expf(-s));
        out[(long)(blockIdx.x * 16 + row) * NUM_CLASSES + cls] = sig;
    }
}

// ---------------------------------------------------------------------------
extern "C" void kernel_launch(void* const* d_in, const int* in_sizes, int n_in,
                              void* d_out, int out_size, void* d_ws, size_t ws_size,
                              hipStream_t stream) {
    const float*     in_feat = (const float*)    d_in[0];
    const long long* src     = (const long long*)d_in[1];
    const long long* dst     = (const long long*)d_in[2];
    const float*     W_self  = (const float*)    d_in[3];
    const float*     W_neigh = (const float*)    d_in[4];
    const float*     b_neigh = (const float*)    d_in[5];
    const float*     W_fc    = (const float*)    d_in[6];
    const float*     b_fc    = (const float*)    d_in[7];
    float* out = (float*)d_out;

    // Workspace layout (≈25.9 MB):
    //   summed : N_NODES*F f32
    //   deg    : N_NODES   f32
    //   wbf    : 2*F*F     bf16  (W_self | W_neigh), 32B-aligned
    float*  summed = (float*)d_ws;
    float*  deg    = summed + (size_t)N_NODES * F;
    __bf16* wbf    = (__bf16*)(deg + N_NODES);

    const int nzero = N_NODES * F + N_NODES;
    sage_zero_ws<<<2048, 256, 0, stream>>>(summed, nzero);

    sage_convert_weights<<<64, 256, 0, stream>>>(W_self, W_neigh, wbf);

    sage_edge_scatter<<<2048, 256, 0, stream>>>(in_feat, src, dst, summed, deg);

    sage_gemm_fused<<<N_NODES / 16, 256, 0, stream>>>(
        in_feat, summed, deg, wbf, b_neigh, W_fc, b_fc, out);
}
